// LSTM_34883724378521
// MI455X (gfx1250) — matrix operand order repaired and checked by reading, not compiled, not run on hardware
//
#include <hip/hip_runtime.h>
#include <cstdint>
#include <cstddef>

// ---------------------------------------------------------------------------
// LSTM (T=60 fed + 54 autoregressive), H=C=128, N=4096, fp32 in/out.
// Strategy: per-sample recurrence is independent -> persistent workgroups,
// each owning 32 rows of N for all 114 steps. Weights (bf16) live in LDS in
// pre-swizzled WMMA B-fragment layout; gates computed with
// v_wmma_f32_16x16x32_bf16 (fp32 accumulate).
// ---------------------------------------------------------------------------

typedef __attribute__((ext_vector_type(16))) __bf16 v16bf;
typedef __attribute__((ext_vector_type(8)))  float  v8f;

#define T_STEPS  60
#define CH       128
#define NPIX     4096
#define HID      128
#define ROWS     32                 // N-rows per workgroup
#define THREADS  256                // 8 wave32
#define ABS      264                // abuf row stride (ushorts): 256 data + 8 pad
#define WFRAG_B  (256 * 1024)       // 256 fragments x 1KB = 256 KB
#define SMEM_BYTES (WFRAG_B + ROWS * ABS * 2)

__device__ __forceinline__ unsigned short f2bf(float f) {
  unsigned int u = __float_as_uint(f);
  unsigned int r = (u + 0x7FFFu + ((u >> 16) & 1u)) >> 16;  // RNE
  return (unsigned short)r;
}
__device__ __forceinline__ float sigmoidf(float x) {
  return 1.0f / (1.0f + __expf(-x));
}

union Frag { v16bf v; uint4 q[2]; };

__global__ __launch_bounds__(THREADS)
void lstm_wmma_kernel(const float* __restrict__ img,
                      const float* __restrict__ Wih,
                      const float* __restrict__ Whh,
                      const float* __restrict__ bih,
                      const float* __restrict__ bhh,
                      float* __restrict__ out,
                      int TOT) {
  extern __shared__ char smem[];
  unsigned short* wfrag = (unsigned short*)smem;              // 256 KB weight frags
  unsigned short* abuf  = (unsigned short*)(smem + WFRAG_B);  // [ROWS][ABS] bf16: x|h

  const int tid  = threadIdx.x;
  const int lane = tid & 31;
  const int wv   = tid >> 5;   // wave 0..7
  const int mt   = wv >> 2;    // m-tile (16 rows each)
  const int cw   = wv & 3;     // 32-wide h-column chunk
  const int n0   = blockIdx.x * ROWS;

  // ---- Pack [W_ih | W_hh] (K=256 stacked) into LDS as bf16 B-fragments ----
  // Fragment f = jt*8 + kb covers B rows k=kb*32..+31, cols j=jt*16..+15.
  // Lane L holds 16 values (32B): n=L%16, kgrp=(L<16?0:16), dword v -> k=kgrp+2v{,+1}.
  for (int d = tid; d < 256 * 256; d += THREADS) {            // dwords
    int f   = d >> 8;
    int idx = d & 255;
    int fl  = idx >> 3;                                       // lane within fragment
    int v   = idx & 7;                                        // dword within lane
    int jt  = f >> 3, kb = f & 7;
    int j   = jt * 16 + (fl & 15);
    int k0  = kb * 32 + ((fl < 16) ? 0 : 16) + 2 * v;
    float f0, f1;
    if (k0 < CH) { f0 = Wih[j * CH + k0];        f1 = Wih[j * CH + k0 + 1]; }
    else         { f0 = Whh[j * HID + k0 - CH];  f1 = Whh[j * HID + k0 - CH + 1]; }
    ((unsigned int*)wfrag)[d] = (unsigned int)f2bf(f0) | ((unsigned int)f2bf(f1) << 16);
  }
  // Zero activation buffer (h half must start at 0)
  for (int d = tid; d < ROWS * ABS / 2; d += THREADS)
    ((unsigned int*)abuf)[d] = 0u;

  // Bias per accumulator tile: u = gate*2 + c2
  float bsum[8];
#pragma unroll
  for (int u = 0; u < 8; ++u) {
    int g = u >> 1, c2 = u & 1;
    int j = g * 128 + cw * 32 + c2 * 16 + (lane & 15);
    bsum[u] = bih[j] + bhh[j];
  }

  float cst[16];                                  // cell state, [c2][r]
#pragma unroll
  for (int i = 0; i < 16; ++i) cst[i] = 0.0f;

  const int lc = tid >> 1;                        // channel this thread stages
  const int lh = (tid & 1) * 16;                  // 16-row half
  const size_t outn = (size_t)TOT * HID;

  const unsigned short* arow =
      abuf + (mt * 16 + (lane & 15)) * ABS + ((lane < 16) ? 0 : 16);

  __syncthreads();

  for (int t = 0; t < TOT; ++t) {
    // ---- Stage x_t (teacher-forced phase); AR phase uses mirrored h ----
    if (t < T_STEPS) {
      const float4* s4 = (const float4*)(img + (size_t)t * CH * NPIX
                                             + (size_t)lc * NPIX + n0 + lh);
#pragma unroll
      for (int i4 = 0; i4 < 4; ++i4) {
        float4 x4 = s4[i4];
        abuf[(lh + i4 * 4 + 0) * ABS + lc] = f2bf(x4.x);
        abuf[(lh + i4 * 4 + 1) * ABS + lc] = f2bf(x4.y);
        abuf[(lh + i4 * 4 + 2) * ABS + lc] = f2bf(x4.z);
        abuf[(lh + i4 * 4 + 3) * ABS + lc] = f2bf(x4.w);
      }
      if (t + 1 < T_STEPS)
        __builtin_prefetch(img + (size_t)(t + 1) * CH * NPIX
                               + (size_t)lc * NPIX + n0 + lh, 0, 0);
    }
    __syncthreads();

    // ---- gates = [x|h] @ [W_ih|W_hh]^T : 64 WMMAs per wave ----
    v8f acc[8];
    const v8f vzero = {0.f, 0.f, 0.f, 0.f, 0.f, 0.f, 0.f, 0.f};
#pragma unroll
    for (int u = 0; u < 8; ++u) acc[u] = vzero;

#pragma unroll
    for (int kb = 0; kb < 8; ++kb) {
      Frag A;
      A.q[0] = *(const uint4*)(arow + kb * 32);
      A.q[1] = *(const uint4*)(arow + kb * 32 + 8);
#pragma unroll
      for (int u = 0; u < 8; ++u) {
        int jt = (u >> 1) * 8 + cw * 2 + (u & 1);
        const unsigned short* fb = wfrag + ((jt << 3) + kb) * 512 + lane * 16;
        Frag B;
        B.q[0] = *(const uint4*)(fb);
        B.q[1] = *(const uint4*)(fb + 8);
        acc[u] = __builtin_amdgcn_wmma_f32_16x16x32_bf16(
            false, A.v, false, B.v, (short)0, acc[u], false, false);
      }
    }
    __syncthreads();   // all A/B reads done before h writeback

    // ---- cell update + h writeback + output store ----
#pragma unroll
    for (int c2 = 0; c2 < 2; ++c2) {
#pragma unroll
      for (int r = 0; r < 8; ++r) {
        float iv = sigmoidf(acc[0 + c2][r] + bsum[0 + c2]);
        float fv = sigmoidf(acc[2 + c2][r] + bsum[2 + c2]);
        float gv = tanhf  (acc[4 + c2][r] + bsum[4 + c2]);
        float ov = sigmoidf(acc[6 + c2][r] + bsum[6 + c2]);
        float cn = fv * cst[c2 * 8 + r] + iv * gv;
        cst[c2 * 8 + r] = cn;
        float hv = ov * tanhf(cn);

        int mloc = mt * 16 + r + ((lane < 16) ? 0 : 8);   // row within WG
        int col  = cw * 32 + c2 * 16 + (lane & 15);       // hidden column
        out[(size_t)(n0 + mloc) * outn + (size_t)t * HID + col] = hv;

        unsigned short hb = f2bf(hv);
        abuf[mloc * ABS + CH + col] = hb;                 // h half
        if (t + 1 >= T_STEPS)
          abuf[mloc * ABS + col] = hb;                    // mirror as next x (AR)
      }
    }
  }
}

extern "C" void kernel_launch(void* const* d_in, const int* in_sizes, int n_in,
                              void* d_out, int out_size, void* d_ws, size_t ws_size,
                              hipStream_t stream) {
  const float* img = (const float*)d_in[0];
  // d_in[1] = mask (unused by reference forward)
  const float* Wih = (const float*)d_in[2];
  const float* Whh = (const float*)d_in[3];
  const float* bih = (const float*)d_in[4];
  const float* bhh = (const float*)d_in[5];
  float* out = (float*)d_out;

  int TOT = out_size / (NPIX * HID);   // T + (T - gp - 1) = 114
  dim3 grid(NPIX / ROWS);              // 128 persistent workgroups
  lstm_wmma_kernel<<<grid, THREADS, SMEM_BYTES, stream>>>(
      img, Wih, Whh, bih, bhh, out, TOT);
}
